// VectorizedPairwiseAttentionStructModule_53137335386464
// MI455X (gfx1250) — compile-verified, matrix operand order checked by reading, not run on hardware
//
#include <hip/hip_runtime.h>
#include <hip/hip_bf16.h>

typedef __attribute__((ext_vector_type(16))) _Float16 v16h;
typedef __attribute__((ext_vector_type(8)))  float    v8f;

// Problem constants (match the reference)
#define BB     16
#define N_PER  512
#define M_PER  64
#define VE_DIM 256
#define LE_DIM 128
#define HH     128
#define H2     64
#define SCALE_F 1.0f

// Workspace layout (float units)
#define OFF_WCV 0u                     // 256*128 combined Wv@W1v
#define OFF_WCL 32768u                 // 128*128 combined Wl@W1l
#define OFF_CB  49152u                 // 128 combined bias
#define OFF_BM  49280u                 // 1024*128 ligand-side rows (bias folded)
#define OFF_AV  180352u                // 8192*128 virtual-side rows
#define OFF_W2H 1228928u               // 8192 halfs (4096 float slots) packed W2

// f16 A-operand K mapping for v_wmma_f32_16x16x32_f16 (per ISA 7.12.2):
// lanes 0-15 : i<8 -> K=i,      i>=8 -> K=i+8   (K in {0..7,16..23})
// lanes 16-31: i<8 -> K=i+8,    i>=8 -> K=i+16  (K in {8..15,24..31})
__device__ __host__ inline int kmap(int laneHi, int i) {
    int k = (i < 8) ? i : (i + 8);
    return k + (laneHi ? 8 : 0);
}

// ---------------- precompute kernels ----------------

// Wc_v[r][c] = sum_h Wv[r][h] * W1[(H+h)][c]      (256x128)
__global__ void k_combine_v(const float* __restrict__ Wv, const float* __restrict__ W1,
                            float* __restrict__ Wcv) {
    int r = blockIdx.x, c = threadIdx.x;
    float acc = 0.f;
#pragma unroll 8
    for (int h = 0; h < HH; ++h)
        acc += Wv[r * HH + h] * W1[(HH + h) * HH + c];
    Wcv[r * HH + c] = acc;
}

// Wc_l[r][c] = sum_h Wl[r][h] * W1[h][c]          (128x128)
__global__ void k_combine_l(const float* __restrict__ Wl, const float* __restrict__ W1,
                            float* __restrict__ Wcl) {
    int r = blockIdx.x, c = threadIdx.x;
    float acc = 0.f;
#pragma unroll 8
    for (int h = 0; h < HH; ++h)
        acc += Wl[r * HH + h] * W1[h * HH + c];
    Wcl[r * HH + c] = acc;
}

// c_bias[c] = b1[c] + sum_h bv[h]*W1v[h][c] + sum_h bl[h]*W1l[h][c]
__global__ void k_bias(const float* __restrict__ bv, const float* __restrict__ bl,
                       const float* __restrict__ W1, const float* __restrict__ b1,
                       float* __restrict__ cb) {
    int c = threadIdx.x;
    float acc = b1[c];
#pragma unroll 8
    for (int h = 0; h < HH; ++h)
        acc += bv[h] * W1[(HH + h) * HH + c] + bl[h] * W1[h * HH + c];
    cb[c] = acc;
}

// Pack W2 (128x64 f32) into the f16 B-operand layout, tiled [kt][nt][lane][i]
__global__ void k_pack_w2(const float* __restrict__ W2, _Float16* __restrict__ W2h) {
    int idx  = blockIdx.x * blockDim.x + threadIdx.x;     // 0..8191
    int i    = idx & 15;
    int lane = (idx >> 4) & 31;
    int nt   = (idx >> 9) & 3;
    int kt   = idx >> 11;
    int K = kt * 32 + kmap(lane >> 4, i);
    int N = nt * 16 + (lane & 15);
    W2h[idx] = (_Float16)W2[K * H2 + N];
}

// A[i][c] = sum_k VE[i][k] * Wcv[k][c]   (8192 rows, K=256)
__global__ void k_proj_v(const float* __restrict__ VEin, const float* __restrict__ Wcv,
                         float* __restrict__ A) {
    int i = blockIdx.x, c = threadIdx.x;
    const float* row = VEin + (size_t)i * VE_DIM;
    float acc = 0.f;
#pragma unroll 4
    for (int k = 0; k < VE_DIM; k += 4) {
        float4 e = *(const float4*)(row + k);
        acc += e.x * Wcv[(k + 0) * HH + c];
        acc += e.y * Wcv[(k + 1) * HH + c];
        acc += e.z * Wcv[(k + 2) * HH + c];
        acc += e.w * Wcv[(k + 3) * HH + c];
    }
    A[(size_t)i * HH + c] = acc;
}

// Bm[j][c] = c_bias[c] + sum_k LE[j][k] * Wcl[k][c]   (1024 rows, K=128)
__global__ void k_proj_l(const float* __restrict__ LEin, const float* __restrict__ Wcl,
                         const float* __restrict__ cb, float* __restrict__ Bm) {
    int j = blockIdx.x, c = threadIdx.x;
    const float* row = LEin + (size_t)j * LE_DIM;
    float acc = cb[c];
#pragma unroll 4
    for (int k = 0; k < LE_DIM; k += 4) {
        float4 e = *(const float4*)(row + k);
        acc += e.x * Wcl[(k + 0) * HH + c];
        acc += e.y * Wcl[(k + 1) * HH + c];
        acc += e.z * Wcl[(k + 2) * HH + c];
        acc += e.w * Wcl[(k + 3) * HH + c];
    }
    Bm[(size_t)j * HH + c] = acc;
}

// ---------------- main fused pairwise-MLP + softmax + coords kernel ----------------
// One workgroup per (b, m): 8 waves x 64 n-rows each. WMMA f16 16x16x32 for h @ W2.
__global__ __launch_bounds__(256) void k_pair_attn(
    const float* __restrict__ A,          // [8192][128] virtual-side
    const float* __restrict__ Bm,         // [1024][128] ligand-side (+bias)
    const _Float16* __restrict__ W2h,     // packed f16 B operands
    const float* __restrict__ b2,         // [64]
    const float* __restrict__ W3,         // [64]
    const float* __restrict__ b3,         // [1]
    const float* __restrict__ vcoords,    // [8192][3]
    float* __restrict__ out_coords,       // [1024][3]
    float* __restrict__ out_attn)         // [1024][512]
{
    __shared__ float base_s[HH];
    __shared__ float scores_s[N_PER];
    __shared__ float red_s[256];

    const int tid  = threadIdx.x;
    const int bmid = blockIdx.x;          // 0..1023
    const int b    = bmid >> 6;
    const int lane = tid & 31;
    const int wave = tid >> 5;
    const int hi   = lane >> 4;           // half-wave select
    const int lo   = lane & 15;

    // stage ligand row (+bias) for this (b,m)
    if (tid < HH) base_s[tid] = Bm[(size_t)bmid * HH + tid];
    __syncthreads();

    const float b3v = b3[0];
    const size_t rowbase_g = (size_t)b * N_PER;   // first virtual row of this batch

#pragma unroll 1
    for (int t = 0; t < 4; ++t) {
        const int n0 = wave * 64 + t * 16;        // tile of 16 n-rows
        const float* arow = A + (rowbase_g + n0 + lo) * HH;
        if (t < 3) __builtin_prefetch(arow + 16 * HH, 0, 1);   // global_prefetch_b8

        v8f acc[4] = {};                           // 16x64 f32 accumulators

#pragma unroll
        for (int kt = 0; kt < 4; ++kt) {
            // Build f16 A operand: h = relu(A[n] + base), ISA lane layout
            const float* ak = arow + kt * 32;
            const float* bk = base_s + kt * 32;
            v16h amat;
#pragma unroll
            for (int i = 0; i < 16; ++i) {
                int k = kmap(hi, i);
                float h = ak[k] + bk[k];
                amat[i] = (_Float16)(h > 0.f ? h : 0.f);
            }
#pragma unroll
            for (int nt = 0; nt < 4; ++nt) {
                const v16h bmat = *(const v16h*)(W2h + (((kt * 4 + nt) * 32 + lane) << 4));
                acc[nt] = __builtin_amdgcn_wmma_f32_16x16x32_f16(
                    false, amat, false, bmat, (short)0, acc[nt], false, false);
            }
        }

        // relu(h2 + b2) . W3 : per-lane partial per output row, then 16-lane butterfly
        float p[8];
#pragma unroll
        for (int r = 0; r < 8; ++r) p[r] = 0.f;
#pragma unroll
        for (int nt = 0; nt < 4; ++nt) {
            const int col = nt * 16 + lo;
            const float bb = b2[col];
            const float w3 = W3[col];
#pragma unroll
            for (int r = 0; r < 8; ++r) {
                float v = acc[nt][r] + bb;
                p[r] += (v > 0.f ? v : 0.f) * w3;
            }
        }
#pragma unroll
        for (int mask = 1; mask <= 8; mask <<= 1)
#pragma unroll
            for (int r = 0; r < 8; ++r)
                p[r] += __shfl_xor(p[r], mask, 32);

        if (lo == 0) {
            const int rb = n0 + hi * 8;           // D layout: lanes>=16 hold rows 8..15
#pragma unroll
            for (int r = 0; r < 8; ++r)
                scores_s[rb + r] = (p[r] + b3v) * SCALE_F;
        }
    }
    __syncthreads();

    // ---- softmax over N=512 (each thread owns 2 entries) ----
    const float s1 = scores_s[tid];
    const float s2 = scores_s[tid + 256];
    red_s[tid] = fmaxf(s1, s2);
    __syncthreads();
#pragma unroll
    for (int s = 128; s > 0; s >>= 1) {
        if (tid < s) red_s[tid] = fmaxf(red_s[tid], red_s[tid + s]);
        __syncthreads();
    }
    const float gmax = red_s[0];
    __syncthreads();

    const float e1 = __expf(s1 - gmax);
    const float e2 = __expf(s2 - gmax);
    red_s[tid] = e1 + e2;
    __syncthreads();
#pragma unroll
    for (int s = 128; s > 0; s >>= 1) {
        if (tid < s) red_s[tid] += red_s[tid + s];
        __syncthreads();
    }
    const float inv = 1.f / red_s[0];
    __syncthreads();

    const float a1 = e1 * inv, a2 = e2 * inv;
    out_attn[(size_t)bmid * N_PER + tid]       = a1;
    out_attn[(size_t)bmid * N_PER + tid + 256] = a2;

    // ---- coords = attn @ virtual_coords ----
    const float* vc = vcoords + rowbase_g * 3;
    float cc[3];
#pragma unroll
    for (int c = 0; c < 3; ++c)
        cc[c] = a1 * vc[tid * 3 + c] + a2 * vc[(tid + 256) * 3 + c];
#pragma unroll
    for (int c = 0; c < 3; ++c) {
        red_s[tid] = cc[c];
        __syncthreads();
#pragma unroll
        for (int s = 128; s > 0; s >>= 1) {
            if (tid < s) red_s[tid] += red_s[tid + s];
            __syncthreads();
        }
        if (tid == 0) out_coords[bmid * 3 + c] = red_s[0];
        __syncthreads();
    }
}

extern "C" void kernel_launch(void* const* d_in, const int* in_sizes, int n_in,
                              void* d_out, int out_size, void* d_ws, size_t ws_size,
                              hipStream_t stream) {
    (void)in_sizes; (void)n_in; (void)out_size; (void)ws_size;

    const float* VEin = (const float*)d_in[0];    // virtual_embeddings (8192,256)
    const float* VC   = (const float*)d_in[1];    // virtual_coords     (8192,3)
    const float* LEin = (const float*)d_in[2];    // ligand_embeddings  (1024,128)
    // d_in[3,4,5]: batch indices / mask — structurally fixed, unused
    const float* Wv = (const float*)d_in[6];
    const float* bv = (const float*)d_in[7];
    const float* Wl = (const float*)d_in[8];
    const float* bl = (const float*)d_in[9];
    const float* W1 = (const float*)d_in[10];
    const float* b1 = (const float*)d_in[11];
    const float* W2 = (const float*)d_in[12];
    const float* b2 = (const float*)d_in[13];
    const float* W3 = (const float*)d_in[14];
    const float* b3 = (const float*)d_in[15];

    float* ws = (float*)d_ws;
    float*     Wcv  = ws + OFF_WCV;
    float*     Wcl  = ws + OFF_WCL;
    float*     cb   = ws + OFF_CB;
    float*     BmW  = ws + OFF_BM;
    float*     Aw   = ws + OFF_AV;
    _Float16*  W2h  = (_Float16*)(ws + OFF_W2H);

    float* out_coords = (float*)d_out;
    float* out_attn   = (float*)d_out + BB * M_PER * 3;

    k_combine_v<<<VE_DIM, HH, 0, stream>>>(Wv, W1, Wcv);
    k_combine_l<<<LE_DIM, HH, 0, stream>>>(Wl, W1, Wcl);
    k_bias     <<<1, HH, 0, stream>>>(bv, bl, W1, b1, cb);
    k_pack_w2  <<<32, 256, 0, stream>>>(W2, W2h);
    k_proj_v   <<<BB * N_PER, HH, 0, stream>>>(VEin, Wcv, Aw);
    k_proj_l   <<<BB * M_PER, HH, 0, stream>>>(LEin, Wcl, cb, BmW);

    k_pair_attn<<<BB * M_PER, 256, 0, stream>>>(Aw, BmW, W2h, b2, W3, b3, VC,
                                                out_coords, out_attn);
}
